// DynamicKBasis_80255758893471
// MI455X (gfx1250) — matrix operand order repaired and checked by reading, not compiled, not run on hardware
//
#include <hip/hip_runtime.h>
#include <cstdint>

// ---------------------------------------------------------------------------
// DynamicKBasis on MI455X (gfx1250, wave32)
// B=32768, N=17, D=2, K=3, H=64
// Layer-2 GEMMs run on v_wmma_f32_16x16x32_f16; layer-1 (K=2) is VALU.
// ---------------------------------------------------------------------------

typedef __attribute__((ext_vector_type(16))) _Float16 v16h;
typedef __attribute__((ext_vector_type(8)))  float    v8f;

#define BB 32768
#define NNs 17
#define KKs 3
#define DDs 2

// output segment offsets in the concatenated f32 d_out
#define OFF_W   0
#define OFF_MU  (BB*NNs*KKs)                  // 1,671,168
#define OFF_SIG (OFF_MU + BB*NNs*KKs*DDs)     // 5,013,504
#define OFF_KJS (OFF_SIG + BB*NNs*KKs*DDs)    // 8,355,840
#define OFF_TOP (OFF_KJS + BB*NNs)            // 8,912,896

// Build f16 A fragment (16x32 tile of hidden activations, ISA 16-bit A layout:
// lanes 0-15 -> M, VGPR v pair p -> K = (v>>2)*16 + half*8 + (v&3)*2 + p).
// h[j] = relu(x0*W1[0][j] + x1*W1[1][j] + b1[j]);  W1 in LDS as [d*H + j].
__device__ __forceinline__ v16h build_a(int lane, int kbase, float x0, float x1,
                                        const float* __restrict__ W1,
                                        const float* __restrict__ b1, int H) {
  const int halfq = lane >> 4;
  v16h a;
#pragma unroll
  for (int v = 0; v < 8; ++v) {
#pragma unroll
    for (int p = 0; p < 2; ++p) {
      const int j = kbase + ((v >> 2) << 4) + (halfq << 3) + ((v & 3) << 1) + p;
      float h = fmaf(x0, W1[j], fmaf(x1, W1[H + j], b1[j]));
      h = fmaxf(h, 0.0f);
      a[2 * v + p] = (_Float16)h;
    }
  }
  return a;
}

// Build f16 B fragment (32x16 tile of layer-2 weights, padded to 16 cols).
// ISA 16-bit B layout: lane l holds column N = l%16; element e holds
// K = kbase + (e>>3)*16 + (l>>4)*8 + (e&7).
__device__ __forceinline__ v16h build_b(int lane, int kbase,
                                        const float* __restrict__ W2p) {
  const int nc = lane & 15, hq = lane >> 4;
  v16h b;
#pragma unroll
  for (int e = 0; e < 16; ++e) b[e] = (_Float16)W2p[(kbase + ((e >> 3) << 4) + (hq << 3) + (e & 7)) * 16 + nc];
  return b;
}

__global__ __launch_bounds__(256) void DynamicKBasis_80255758893471_kernel(
    const float* __restrict__ x,
    const float* __restrict__ kW1, const float* __restrict__ kb1,
    const float* __restrict__ kW2, const float* __restrict__ kb2,
    const float* __restrict__ wW1, const float* __restrict__ wb1,
    const float* __restrict__ wW2, const float* __restrict__ wb2,
    const float* __restrict__ bW1, const float* __restrict__ bb1,
    const float* __restrict__ bW2, const float* __restrict__ bb2,
    float* __restrict__ o_w, float* __restrict__ o_mu,
    float* __restrict__ o_sig, float* __restrict__ o_kjs,
    float* __restrict__ o_top) {
  const int n    = blockIdx.y;
  const int t    = threadIdx.x;
  const int lane = t & 31;
  const int wave = t >> 5;
  const int halfq = lane >> 4;

  __shared__ float s_bW2p[128 * 16];   // padded 12 -> 16 cols
  __shared__ float s_kW2p[64 * 16];    // padded 3 -> 16 cols
  __shared__ float s_wW2p[64 * 16];
  __shared__ float s_bW1[2 * 128];
  __shared__ float s_bb1[128];
  __shared__ float s_bb2[16];
  __shared__ float s_kW1[2 * 64];
  __shared__ float s_kb1[64];
  __shared__ float s_kb2[4];
  __shared__ float s_wW1[2 * 64];
  __shared__ float s_wb1[64];
  __shared__ float s_wb2[4];
  __shared__ float s_spill[8][16][8];  // [wave][row m][0..2 k-logit | 4..6 w-logit]

  // ---- stage this n's weights into LDS (cooperative) ----
  for (int i = t; i < 128 * 16; i += 256) {
    const int r = i >> 4, c = i & 15;
    s_bW2p[i] = (c < 12) ? bW2[(n * 128 + r) * 12 + c] : 0.0f;
  }
  for (int i = t; i < 64 * 16; i += 256) {
    const int r = i >> 4, c = i & 15;
    s_kW2p[i] = (c < 3) ? kW2[(n * 64 + r) * 3 + c] : 0.0f;
    s_wW2p[i] = (c < 3) ? wW2[(n * 64 + r) * 3 + c] : 0.0f;
  }
  if (t < 256) s_bW1[t] = bW1[n * 256 + t];
  if (t < 128) {
    s_bb1[t] = bb1[n * 128 + t];
    s_kW1[t] = kW1[n * 128 + t];   // (N,2,64) flat
    s_wW1[t] = wW1[n * 128 + t];
  }
  if (t < 64) { s_kb1[t] = kb1[n * 64 + t]; s_wb1[t] = wb1[n * 64 + t]; }
  if (t < 16) s_bb2[t] = (t < 12) ? bb2[n * 12 + t] : 0.0f;
  if (t < 4)  { s_kb2[t] = (t < 3) ? kb2[n * 3 + t] : 0.0f;
                s_wb2[t] = (t < 3) ? wb2[n * 3 + t] : 0.0f; }
  __syncthreads();

  // ---- B fragments: built once per block, reused across 4 batch tiles ----
  const v16h Bb0 = build_b(lane,  0, s_bW2p), Bb1 = build_b(lane, 32, s_bW2p);
  const v16h Bb2 = build_b(lane, 64, s_bW2p), Bb3 = build_b(lane, 96, s_bW2p);
  const v16h Bk0 = build_b(lane,  0, s_kW2p), Bk1 = build_b(lane, 32, s_kW2p);
  const v16h Bw0 = build_b(lane,  0, s_wW2p), Bw1 = build_b(lane, 32, s_wW2p);

  const int gw = blockIdx.x * 8 + wave;          // 0..511 global wave per n
#pragma unroll 1
  for (int it = 0; it < 4; ++it) {
    const int tile = gw * 4 + it;                // 0..2047
    const int b0   = tile * 16;
    const int brow = b0 + (lane & 15);
    const float x0 = x[(brow * NNs + n) * 2 + 0];
    const float x1 = x[(brow * NNs + n) * 2 + 1];

    // ---- k-MLP layer2 on WMMA (K=64, N padded to 16) ----
    v8f ck = {};
    {
      v16h a = build_a(lane, 0, x0, x1, s_kW1, s_kb1, 64);
      ck = __builtin_amdgcn_wmma_f32_16x16x32_f16(false, a, false, Bk0, (short)0, ck, false, false);
      a = build_a(lane, 32, x0, x1, s_kW1, s_kb1, 64);
      ck = __builtin_amdgcn_wmma_f32_16x16x32_f16(false, a, false, Bk1, (short)0, ck, false, false);
    }
    // ---- w-MLP layer2 on WMMA ----
    v8f cw = {};
    {
      v16h a = build_a(lane, 0, x0, x1, s_wW1, s_wb1, 64);
      cw = __builtin_amdgcn_wmma_f32_16x16x32_f16(false, a, false, Bw0, (short)0, cw, false, false);
      a = build_a(lane, 32, x0, x1, s_wW1, s_wb1, 64);
      cw = __builtin_amdgcn_wmma_f32_16x16x32_f16(false, a, false, Bw1, (short)0, cw, false, false);
    }

    // spill the 3 valid logit columns so each row's logits land in one lane
    const int nn = lane & 15;
    if (nn < 3) {
#pragma unroll
      for (int i = 0; i < 8; ++i) {
        const int m = halfq * 8 + i;             // C layout: lanes16-31 -> M=8+i
        s_spill[wave][m][nn]     = ck[i] + s_kb2[nn];
        s_spill[wave][m][4 + nn] = cw[i] + s_wb2[nn];
      }
    }
    asm volatile("s_wait_dscnt 0" ::: "memory"); // intra-wave DS ordering

    {
      const int m  = lane & 15;
      const int b  = b0 + m;
      const int bn = b * NNs + n;
      const float L0 = s_spill[wave][m][halfq * 4 + 0];
      const float L1 = s_spill[wave][m][halfq * 4 + 1];
      const float L2 = s_spill[wave][m][halfq * 4 + 2];
      if (halfq == 0) {
        // argmax (first max wins, like jnp.argmax) -> k_js = idx+1
        int   am = (L1 > L0) ? 1 : 0;
        float bv = (L1 > L0) ? L1 : L0;
        if (L2 > bv) am = 2;
        o_kjs[bn] = (float)(am + 1);
      } else {
        // softmax + stable descending argsort over 3 values
        const float mx = fmaxf(L0, fmaxf(L1, L2));
        const float e0 = __expf(L0 - mx), e1 = __expf(L1 - mx), e2 = __expf(L2 - mx);
        const float inv = 1.0f / (e0 + e1 + e2);
        const float w0 = e0 * inv, w1 = e1 * inv, w2 = e2 * inv;
        o_w[bn * 3 + 0] = w0; o_w[bn * 3 + 1] = w1; o_w[bn * 3 + 2] = w2;
        const int g0 = (w1 >  w0) + (w2 >  w0);
        const int g1 = (w0 >= w1) + (w2 >  w1);
        const int g2 = (w0 >= w2) + (w1 >= w2);
        o_top[bn * 3 + g0] = 0.0f;
        o_top[bn * 3 + g1] = 1.0f;
        o_top[bn * 3 + g2] = 2.0f;
      }
    }

    // ---- b-MLP layer2 on WMMA (K=128, N padded 12->16) ----
    v8f cb = {};
    {
      v16h a = build_a(lane, 0, x0, x1, s_bW1, s_bb1, 128);
      cb = __builtin_amdgcn_wmma_f32_16x16x32_f16(false, a, false, Bb0, (short)0, cb, false, false);
      a = build_a(lane, 32, x0, x1, s_bW1, s_bb1, 128);
      cb = __builtin_amdgcn_wmma_f32_16x16x32_f16(false, a, false, Bb1, (short)0, cb, false, false);
      a = build_a(lane, 64, x0, x1, s_bW1, s_bb1, 128);
      cb = __builtin_amdgcn_wmma_f32_16x16x32_f16(false, a, false, Bb2, (short)0, cb, false, false);
      a = build_a(lane, 96, x0, x1, s_bW1, s_bb1, 128);
      cb = __builtin_amdgcn_wmma_f32_16x16x32_f16(false, a, false, Bb3, (short)0, cb, false, false);
    }
    if (nn < 12) {
      const int k  = nn >> 2;
      const int tt = nn & 3;
#pragma unroll
      for (int i = 0; i < 8; ++i) {
        const int m = halfq * 8 + i;
        const int b = b0 + m;
        const size_t base = ((size_t)(b * NNs + n) * KKs + k) * 2;
        const float val = cb[i] + s_bb2[nn];
        if (tt < 2) o_mu[base + tt] = val;          // params[..., :D]
        else        o_sig[base + (tt - 2)] = __expf(val);  // exp(params[..., D:])
      }
    }
  }
}

extern "C" void kernel_launch(void* const* d_in, const int* in_sizes, int n_in,
                              void* d_out, int out_size, void* d_ws, size_t ws_size,
                              hipStream_t stream) {
  (void)in_sizes; (void)n_in; (void)out_size; (void)d_ws; (void)ws_size;
  const float* x   = (const float*)d_in[0];
  const float* kW1 = (const float*)d_in[1];
  const float* kb1 = (const float*)d_in[2];
  const float* kW2 = (const float*)d_in[3];
  const float* kb2 = (const float*)d_in[4];
  const float* wW1 = (const float*)d_in[5];
  const float* wb1 = (const float*)d_in[6];
  const float* wW2 = (const float*)d_in[7];
  const float* wb2 = (const float*)d_in[8];
  const float* bW1 = (const float*)d_in[9];
  const float* bb1 = (const float*)d_in[10];
  const float* bW2 = (const float*)d_in[11];
  const float* bb2 = (const float*)d_in[12];

  float* out   = (float*)d_out;
  float* o_w   = out + OFF_W;
  float* o_mu  = out + OFF_MU;
  float* o_sig = out + OFF_SIG;
  float* o_kjs = out + OFF_KJS;
  float* o_top = out + OFF_TOP;

  dim3 grid(64, NNs, 1);   // 64 blocks * 8 waves * 4 tiles * 16 rows = 32768
  dim3 block(256, 1, 1);
  DynamicKBasis_80255758893471_kernel<<<grid, block, 0, stream>>>(
      x, kW1, kb1, kW2, kb2, wW1, wb1, wW2, wb2, bW1, bb1, bW2, bb2,
      o_w, o_mu, o_sig, o_kjs, o_top);
}
